// LSTM_index_72516227825842
// MI455X (gfx1250) — matrix operand order repaired and checked by reading, not compiled, not run on hardware
//
#include <hip/hip_runtime.h>
#include <hip/hip_bf16.h>

#define LSTM 256
#define NB 6
#define NSLOT 7

typedef __attribute__((ext_vector_type(16))) __bf16 bf16x16;
typedef __attribute__((ext_vector_type(8))) float f32x8;

__device__ __forceinline__ unsigned short f2bf(float f) {
  unsigned int u = __float_as_uint(f);
  u += 0x7FFFu + ((u >> 16) & 1u);
  return (unsigned short)(u >> 16);
}
__device__ __forceinline__ float bf2f(unsigned short u) {
  return __uint_as_float(((unsigned int)u) << 16);
}

// ---- fast transcendentals: gfx1250 has V_TANH_F32 / V_EXP_F32 / V_LOG_F32 TRANS ops
__device__ __forceinline__ float fast_tanh(float x) {
#if __has_builtin(__builtin_amdgcn_tanhf)
  return __builtin_amdgcn_tanhf(x);
#else
  return tanhf(x);
#endif
}
__device__ __forceinline__ float fast_exp2(float x) {
#if __has_builtin(__builtin_amdgcn_exp2f)
  return __builtin_amdgcn_exp2f(x);
#else
  return exp2f(x);
#endif
}
__device__ __forceinline__ float fast_rcp(float x) {
#if __has_builtin(__builtin_amdgcn_rcpf)
  return __builtin_amdgcn_rcpf(x);
#else
  return 1.0f / x;
#endif
}
__device__ __forceinline__ float fast_sigmoid(float x) {
  // 1/(1+exp(-x)) = 1/(1+exp2(-x*log2e))
  return fast_rcp(1.0f + fast_exp2(x * -1.4426950408889634f));
}

union BfFrag { bf16x16 v; unsigned short u[16]; };

// A fragment: W row-major bf16, tile rows [row_base, row_base+16), K slice [k0, k0+32)
// Contiguous K order used here (same as B): lane m = lane&15; lanes<16 -> K = k0+{0..15}; lanes>=16 -> +16
__device__ __forceinline__ bf16x16 load_a_frag(const unsigned short* __restrict__ W,
                                               int row_base, int k0, int lane) {
  const int m = lane & 15;
  const int kh = (lane >> 4) << 4;
  const unsigned short* p = W + (row_base + m) * LSTM + k0 + kh;
  BfFrag a;
#pragma unroll
  for (int e = 0; e < 16; ++e) a.u[e] = p[e];
  return a.v;
}
// B fragment: column vector contiguous in K; lane n = lane&15, lanes>=16 -> K += 16
__device__ __forceinline__ bf16x16 load_b_frag(const unsigned short* __restrict__ col,
                                               int k0, int lane) {
  const int kb = (lane >> 4) << 4;
  const unsigned short* p = col + k0 + kb;
  BfFrag b;
#pragma unroll
  for (int e = 0; e < 16; ++e) b.u[e] = p[e];
  return b.v;
}
__device__ __forceinline__ f32x8 wmma_bf(bf16x16 a, bf16x16 b, f32x8 c) {
  return __builtin_amdgcn_wmma_f32_16x16x32_bf16(false, a, false, b, (short)0, c, false, false);
}

// Threefry-2x32, 20 rounds (JAX default PRNG)
__device__ __forceinline__ void threefry2x32(unsigned k0, unsigned k1,
                                             unsigned x0, unsigned x1,
                                             unsigned* o0, unsigned* o1) {
  const unsigned R0[4] = {13u, 15u, 26u, 6u};
  const unsigned R1[4] = {17u, 29u, 16u, 24u};
  unsigned ks[3] = {k0, k1, k0 ^ k1 ^ 0x1BD11BDAu};
  x0 += ks[0]; x1 += ks[1];
#pragma unroll
  for (int g = 0; g < 5; ++g) {
    const unsigned* R = (g & 1) ? R1 : R0;
#pragma unroll
    for (int r = 0; r < 4; ++r) {
      x0 += x1;
      x1 = (x1 << R[r]) | (x1 >> (32u - R[r]));
      x1 ^= x0;
    }
    x0 += ks[(g + 1) % 3];
    x1 += ks[(g + 2) % 3] + (unsigned)(g + 1);
  }
  *o0 = x0; *o1 = x1;
}

__device__ __forceinline__ float gumbel_from_bits(unsigned bits) {
  const float tiny = 1.1754943508222875e-38f;
  float f = __uint_as_float((bits >> 9) | 0x3F800000u) - 1.0f;   // [0,1)
  float u = fmaxf(tiny, f * (1.0f - tiny) + tiny);               // jax.uniform(tiny,1)
  return -logf(-logf(u));                                        // libm: sampler fidelity
}

__global__ void enas_prep_kernel(const float* __restrict__ wih, const float* __restrict__ whh,
                                 const float* __restrict__ w1, const float* __restrict__ w2,
                                 const float* __restrict__ bih, const float* __restrict__ bhh,
                                 unsigned short* __restrict__ o_wih, unsigned short* __restrict__ o_whh,
                                 unsigned short* __restrict__ o_w1, unsigned short* __restrict__ o_w2,
                                 float* __restrict__ o_bsum) {
  const int stride = gridDim.x * blockDim.x;
  const int start = blockIdx.x * blockDim.x + threadIdx.x;
  for (int k = start; k < 4 * LSTM * LSTM; k += stride) { o_wih[k] = f2bf(wih[k]); o_whh[k] = f2bf(whh[k]); }
  for (int k = start; k < LSTM * LSTM; k += stride) { o_w1[k] = f2bf(w1[k]); o_w2[k] = f2bf(w2[k]); }
  for (int k = start; k < 4 * LSTM; k += stride) o_bsum[k] = bih[k] + bhh[k];
}

__global__ __launch_bounds__(256) void enas_sampler_kernel(
    const float* __restrict__ enc_w, const float* __restrict__ v_vec,
    const unsigned short* __restrict__ Wih, const unsigned short* __restrict__ Whh,
    const unsigned short* __restrict__ W1, const unsigned short* __restrict__ W2,
    const float* __restrict__ bsum, float* __restrict__ out) {
  __shared__ unsigned short Xbf[NB][LSTM];          // inputs (bf16), column per block
  __shared__ unsigned short Hbf[2][NB][LSTM];       // h state ping-pong (bf16)
  __shared__ float cF[NB][LSTM];                    // c state (f32)
  __shared__ unsigned short anchBf[NSLOT][NB][LSTM];
  __shared__ unsigned short aw1Bf[NSLOT][NB][LSTM]; // w1 @ h anchors (bf16)
  __shared__ float w2hF[NB][LSTM];
  __shared__ unsigned short enc0bf[LSTM];
  __shared__ float logitsF[NB][8];
  __shared__ int idxSel[NB];
  __shared__ float lpAcc[NB];
  __shared__ float entAcc[NB];
  __shared__ unsigned bk0[NB], bk1[NB];

  const int tid = threadIdx.x;
  const int lane = tid & 31;
  const int wave = tid >> 5;
  const int ncol = lane & 15;                 // D/B column = block id
  const int nclamp = (ncol < NB) ? ncol : 0;  // pad columns read col 0
  const int mh8 = (lane >> 4) * 8;            // D row offset for upper lane half

  for (int k = tid; k < LSTM; k += 256) enc0bf[k] = f2bf(enc_w[k]);
  for (int i = tid; i < NB * LSTM; i += 256) {
    const int b = i >> 8, k = i & 255;
    cF[b][k] = 0.0f;
    Hbf[0][b][k] = 0; Hbf[1][b][k] = 0;
    anchBf[0][b][k] = 0; anchBf[1][b][k] = 0;   // anchors[0],[1] are zeros
  }
  if (tid < NB) {
    lpAcc[tid] = 0.0f; entAcc[tid] = 0.0f;
    unsigned a0, a1;
    threefry2x32(0u, 42u, 0u, (unsigned)tid, &a0, &a1);  // fold_in(key(42), b)
    bk0[tid] = a0; bk1[tid] = a1;
  }
  __syncthreads();
  for (int i = tid; i < NB * LSTM; i += 256) Xbf[i >> 8][i & 255] = enc0bf[i & 255];
  __syncthreads();

  int cur = 0;
  const f32x8 zero8 = {0, 0, 0, 0, 0, 0, 0, 0};

  // One LSTM cell for all 6 blocks in lockstep: gates[1024x16] via WMMA, fused nonlinearity.
  auto cell = [&](bool zeroState, int anchorSlot, bool writeOut) {
    const unsigned short* xcol = &Xbf[nclamp][0];
    const unsigned short* hcol = &Hbf[cur][nclamp][0];
#pragma unroll
    for (int tt = 0; tt < 2; ++tt) {
      const int t = wave + tt * 8;  // hidden tile (16 rows)
      f32x8 acc[4];
#pragma unroll
      for (int g = 0; g < 4; ++g) acc[g] = zero8;
#pragma unroll 2
      for (int kk = 0; kk < 8; ++kk) {
        const int k0 = kk * 32;
        bf16x16 bx = load_b_frag(xcol, k0, lane);
#pragma unroll
        for (int g = 0; g < 4; ++g) {
          bf16x16 a = load_a_frag(Wih, g * 256 + t * 16, k0, lane);
          acc[g] = wmma_bf(a, bx, acc[g]);
        }
      }
      if (!zeroState) {
#pragma unroll 2
        for (int kk = 0; kk < 8; ++kk) {
          const int k0 = kk * 32;
          bf16x16 bh = load_b_frag(hcol, k0, lane);
#pragma unroll
          for (int g = 0; g < 4; ++g) {
            bf16x16 a = load_a_frag(Whh, g * 256 + t * 16, k0, lane);
            acc[g] = wmma_bf(a, bh, acc[g]);
          }
        }
      }
#pragma unroll
      for (int r = 0; r < 8; ++r) {
        const int j = t * 16 + mh8 + r;
        const float gi = acc[0][r] + bsum[j];
        const float gf = acc[1][r] + bsum[256 + j];
        const float gg = acc[2][r] + bsum[512 + j];
        const float go = acc[3][r] + bsum[768 + j];
        if (ncol < NB) {
          const float cold = zeroState ? 0.0f : cF[ncol][j];
          const float si = fast_sigmoid(gi);
          const float sf = fast_sigmoid(gf);
          const float so = fast_sigmoid(go);
          const float c2 = sf * cold + si * fast_tanh(gg);
          const float h2 = so * fast_tanh(c2);
          cF[ncol][j] = c2;
          const unsigned short hb = f2bf(h2);
          Hbf[cur ^ 1][ncol][j] = hb;
          if (anchorSlot >= 0) anchBf[anchorSlot][ncol][j] = hb;
          if (writeOut && ncol == NB - 1) { out[62 + j] = c2; out[62 + 256 + j] = h2; }
        }
      }
    }
    __syncthreads();
  };

  // 256x256 matmul against current h; slot>=0 -> aw1Bf[slot] (bf16), slot<0 -> w2hF (f32)
  auto matvec = [&](const unsigned short* W, int slot) {
    const unsigned short* hcol = &Hbf[cur][nclamp][0];
#pragma unroll
    for (int tt = 0; tt < 2; ++tt) {
      const int t = wave + tt * 8;
      f32x8 acc = zero8;
#pragma unroll 2
      for (int kk = 0; kk < 8; ++kk) {
        const int k0 = kk * 32;
        bf16x16 bh = load_b_frag(hcol, k0, lane);
        bf16x16 a = load_a_frag(W, t * 16, k0, lane);
        acc = wmma_bf(a, bh, acc);
      }
#pragma unroll
      for (int r = 0; r < 8; ++r) {
        const int j = t * 16 + mh8 + r;
        if (ncol < NB) {
          if (slot >= 0) aw1Bf[slot][ncol][j] = f2bf(acc[r]);
          else w2hF[ncol][j] = acc[r];
        }
      }
    }
    __syncthreads();
  };

  // Two warm-up cells: reference passes explicit zero (h, c)
  cell(true, -1, false); cur ^= 1;
  matvec(W1, 0);
  cell(true, -1, false); cur ^= 1;
  matvec(W1, 1);

  int step = 0;
  for (int layer = 2; layer <= 6; ++layer) {
    for (int s2 = 0; s2 < 2; ++s2) {
      cell(false, -1, false); cur ^= 1;
      matvec(W2, -1);
      // logits[l] = 1.1*tanh( dot(tanh(aw1[l]+w2h), v) / 5 )  per block
      {
        const int ntask = NB * layer;
        for (int task = wave; task < ntask; task += 8) {
          const int b = task / layer;
          const int l = task - b * layer;
          float s = 0.0f;
          for (int k = lane; k < LSTM; k += 32)
            s += fast_tanh(bf2f(aw1Bf[l][b][k]) + w2hF[b][k]) * v_vec[k];
#pragma unroll
          for (int off = 16; off > 0; off >>= 1) s += __shfl_xor(s, off, 32);
          if (lane == 0) logitsF[b][l] = 1.1f * fast_tanh(s * 0.2f);
        }
        __syncthreads();
      }
      // Gumbel-argmax categorical + log-softmax stats (JAX-faithful threefry, libm precision)
      if (tid < NB) {
        unsigned sk0, sk1;
        threefry2x32(bk0[tid], bk1[tid], 0u, (unsigned)step, &sk0, &sk1);  // fold_in(bkey, step)
        const int n = layer;
        const int half = (n + 1) >> 1;
        float gum[NSLOT];
        for (int jj = 0; jj < half; ++jj) {
          const unsigned x1v = (half + jj < n) ? (unsigned)(half + jj) : 0u;
          unsigned o0, o1;
          threefry2x32(sk0, sk1, (unsigned)jj, x1v, &o0, &o1);
          gum[jj] = gumbel_from_bits(o0);
          if (half + jj < n) gum[half + jj] = gumbel_from_bits(o1);
        }
        int best = 0; float bestv = -3.4e38f, mx = -3.4e38f;
        for (int l = 0; l < n; ++l) {
          const float lg = logitsF[tid][l];
          const float p = lg + gum[l];
          if (p > bestv) { bestv = p; best = l; }
          mx = fmaxf(mx, lg);
        }
        float ssum = 0.0f;
        for (int l = 0; l < n; ++l) ssum += expf(logitsF[tid][l] - mx);
        const float lse = mx + logf(ssum);
        lpAcc[tid] += -(logitsF[tid][best] - lse);
        float ent = 0.0f;
        for (int l = 0; l < n; ++l) {
          const float ls = logitsF[tid][l] - lse;
          ent -= ls * expf(ls);
        }
        entAcc[tid] += ent;
        idxSel[tid] = best;
        out[tid * 10 + step] = (float)best;
      }
      step++;
      __syncthreads();
      // inputs = anchors[index]
      for (int i = tid; i < NB * LSTM; i += 256) {
        const int b = i >> 8, k = i & 255;
        Xbf[b][k] = anchBf[idxSel[b]][b][k];
      }
      __syncthreads();
    }
    // anchor cell; last cell of last layer emits c,h of block 5
    cell(false, layer, layer == 6); cur ^= 1;
    matvec(W1, layer);
    for (int i = tid; i < NB * LSTM; i += 256) Xbf[i >> 8][i & 255] = enc0bf[i & 255];
    __syncthreads();
  }

  if (tid == 0) {
    float lp = 0.0f, en = 0.0f;
    for (int b = 0; b < NB; ++b) { lp += lpAcc[b]; en += entAcc[b]; }
    out[60] = lp;
    out[61] = en;
  }
}

extern "C" void kernel_launch(void* const* d_in, const int* in_sizes, int n_in,
                              void* d_out, int out_size, void* d_ws, size_t ws_size,
                              hipStream_t stream) {
  const float* enc_w = (const float*)d_in[0];
  const float* w_ih  = (const float*)d_in[1];
  const float* w_hh  = (const float*)d_in[2];
  const float* b_ih  = (const float*)d_in[3];
  const float* b_hh  = (const float*)d_in[4];
  const float* w1    = (const float*)d_in[5];
  const float* w2    = (const float*)d_in[6];
  const float* v     = (const float*)d_in[7];
  (void)in_sizes; (void)n_in; (void)out_size; (void)ws_size;

  unsigned char* ws = (unsigned char*)d_ws;
  size_t off = 0;
  unsigned short* o_wih = (unsigned short*)(ws + off); off += (size_t)4 * LSTM * LSTM * 2;
  unsigned short* o_whh = (unsigned short*)(ws + off); off += (size_t)4 * LSTM * LSTM * 2;
  unsigned short* o_w1  = (unsigned short*)(ws + off); off += (size_t)LSTM * LSTM * 2;
  unsigned short* o_w2  = (unsigned short*)(ws + off); off += (size_t)LSTM * LSTM * 2;
  float* o_bsum         = (float*)(ws + off);          off += (size_t)4 * LSTM * 4;

  enas_prep_kernel<<<256, 256, 0, stream>>>(w_ih, w_hh, w1, w2, b_ih, b_hh,
                                            o_wih, o_whh, o_w1, o_w2, o_bsum);
  enas_sampler_kernel<<<1, 256, 0, stream>>>(enc_w, v, o_wih, o_whh, o_w1, o_w2, o_bsum,
                                             (float*)d_out);
}
